// MultiHeadTransformer_33277406609762
// MI455X (gfx1250) — matrix operand order repaired and checked by reading, not compiled, not run on hardware
//
#include <hip/hip_runtime.h>

// ---------------------------------------------------------------------------
// MultiHeadTransformer block for MI455X (gfx1250, wave32, WMMA + TDM).
// B=2, S=2048, E=1024, H=16, DH=64, FF=4096.  fp32 I/O, bf16 tensor-core math.
//
// Strategy (compute-bound, ~172 GFLOP vs ~µs of HBM time at 23.3 TB/s):
//  * one-time fp32->bf16 conversion of x and all weights (weights transposed
//    to [N][K] so WMMA B-fragments are contiguous in LDS)
//  * all GEMMs + both attention einsums on v_wmma_f32_16x16x32_bf16
//  * LDS tiles staged by the Tensor Data Mover (tensor_load_to_lds) with
//    double buffering + s_wait_tensorcnt; TDM pad fields inject the +8 bf16
//    row padding used for bank-conflict avoidance
// ---------------------------------------------------------------------------

#define B_    2
#define S_    2048
#define E_    1024
#define H_    16
#define DH_   64
#define FF_   4096
#define M_TOT (B_ * S_)      // 4096 rows for all GEMMs
#define LN_EPS 1e-5f

typedef __attribute__((ext_vector_type(16))) __bf16 v16bf;
typedef __attribute__((ext_vector_type(8)))  float  v8f;
typedef __attribute__((ext_vector_type(4)))  unsigned int uv4;
typedef __attribute__((ext_vector_type(8)))  int          iv8;
typedef __attribute__((ext_vector_type(4)))  int          iv4;

// ---------------------------------------------------------------------------
// TDM: issue a 2D tile load (bf16) global -> LDS.  D# layout per ISA
// CDNA5 ISA.  pad_interval/pad_amount are the encoded
// 3b/7b fields (pad after every 2<<i dwords, pad by (a+1) dwords).
// Must be executed by a single wave (uniform args); tracked by TENSORcnt.
// This toolchain exposes the 6-arg builtin (g0, g1, g2, g3, g4, cpol).
// ---------------------------------------------------------------------------
__device__ __forceinline__ void tdm_load_2d_bf16(unsigned lds_byte_addr,
                                                 const void* gaddr,
                                                 int tile_k, int tile_rows,
                                                 int row_stride_elems,
                                                 int pad_interval_code,
                                                 int pad_amount_code) {
  const unsigned long long ga = (unsigned long long)gaddr;
  uv4 g0;
  g0[0] = 1u;                                      // count=1 (valid user D#)
  g0[1] = lds_byte_addr;                           // lds_addr
  g0[2] = (unsigned)(ga & 0xFFFFFFFFu);            // global_addr[31:0]
  g0[3] = (unsigned)((ga >> 32) & 0x1FFFFFFu)      // global_addr[56:32]
        | (2u << 30);                              // type=2 ("image")
  iv8 g1;
  g1[0] = (1 << 16)                                // data_size = 2 bytes
        | (1 << 20)                                // pad_enable
        | (pad_interval_code << 22)
        | (pad_amount_code << 25);
  g1[1] = 0;                                       // td0[15:0]=0 (td0 = 1<<30)
  g1[2] = 0x4000;                                  // td0[31:16] ; td1 lo = 0
  g1[3] = 0x4000 | (tile_k << 16);                 // td1 hi ; tile_dim0
  g1[4] = tile_rows & 0xFFFF;                      // tile_dim1 ; tile_dim2=0
  g1[5] = row_stride_elems;                        // tensor_dim0_stride[31:0]
  g1[6] = 0;
  g1[7] = 0;
  const iv4 z4 = {0, 0, 0, 0};
  const iv8 z8 = {0, 0, 0, 0, 0, 0, 0, 0};
  __builtin_amdgcn_tensor_load_to_lds(g0, g1, z4, z4, z8, 0);
}

__device__ __forceinline__ unsigned lds_addr_of(const void* p) {
  return (unsigned)(unsigned long long)p;   // low 32 bits = LDS byte offset
}

// ---- CDNA5 WMMA operand loaders (layouts per CDNA5 ISA §7.12.2) ------
// A (16x32 bf16): lane L holds row M = L&15; VGPR v holds K pair
//   K = (v&4 ? 16 : 0) + (v&3)*2 + (L>=16 ? 8 : 0)
__device__ __forceinline__ v16bf lds_frag_a(const __bf16* base, int pitch,
                                            int row0, int k0) {
  const int lane = threadIdx.x & 31;
  const int grp  = lane >> 4;
  const __bf16* p = base + (row0 + (lane & 15)) * pitch + k0 + grp * 8;
  v16bf a;
#pragma unroll
  for (int v = 0; v < 8; ++v) {
    const int k = ((v & 4) << 2) + (v & 3) * 2;    // 0,2,4,6,16,18,20,22
    a[2 * v]     = p[k];
    a[2 * v + 1] = p[k + 1];
  }
  return a;
}

// B (32x16 bf16) staged transposed in LDS as [N][K]: lane L holds column
// N = L&15; element e holds K = (e&8 ? 16 : 0) + (e&7) + (L>=16 ? 8 : 0).
__device__ __forceinline__ v16bf lds_frag_b(const __bf16* baseT, int pitch,
                                            int n0, int k0) {
  const int lane = threadIdx.x & 31;
  const int grp  = lane >> 4;
  const __bf16* p = baseT + (n0 + (lane & 15)) * pitch + k0 + grp * 8;
  v16bf b;
#pragma unroll
  for (int e = 0; e < 16; ++e) b[e] = p[e + (e & 8)];
  return b;
}

// ---------------------------------------------------------------------------
// One-time fp32 -> bf16 conversion (vectorized).
// ---------------------------------------------------------------------------
__launch_bounds__(256)
__global__ void convert_bf16(const float* __restrict__ in,
                             __bf16* __restrict__ out, int n) {
  const int i = (blockIdx.x * 256 + threadIdx.x) * 4;
  if (i < n) {
    const float4 v = *(const float4*)(in + i);
    __bf16 o[4] = {(__bf16)v.x, (__bf16)v.y, (__bf16)v.z, (__bf16)v.w};
    __builtin_memcpy(out + i, o, 8);
  }
}

// W[K,N] fp32 -> Wt[N,K] bf16 (32x32 tiles via LDS).
__launch_bounds__(256)
__global__ void transpose_to_bf16(const float* __restrict__ W,
                                  __bf16* __restrict__ Wt, int K, int N) {
  __shared__ float tile[32][33];
  const int k0 = blockIdx.y * 32, n0 = blockIdx.x * 32;
  const int tid = threadIdx.x;
  {
    const int r = tid >> 3, c = (tid & 7) * 4;
    const float4 v = *(const float4*)(W + (size_t)(k0 + r) * N + n0 + c);
    tile[r][c + 0] = v.x; tile[r][c + 1] = v.y;
    tile[r][c + 2] = v.z; tile[r][c + 3] = v.w;
  }
  __syncthreads();
  const int nr = tid >> 3, kc = (tid & 7) * 4;
  __bf16 o[4];
#pragma unroll
  for (int i = 0; i < 4; ++i) o[i] = (__bf16)tile[kc + i][nr];
  __builtin_memcpy(Wt + (size_t)(n0 + nr) * K + k0 + kc, o, 8);
}

// ---------------------------------------------------------------------------
// GEMM: C[M,N] = act(A[M,K] @ W[K,N] + bias[N]) with A bf16 [M,K], weights
// pre-transposed bf16 Wt[N,K].  Block tile 128x128, BK=32, 8 waves (4Mx2N),
// wave tile 32x64 -> 8 WMMAs per K-step.  TDM double-buffered staging.
// M%128==0, N%128==0, K%32==0.
// ---------------------------------------------------------------------------
template <int RELU, int OUT_BF16>
__launch_bounds__(256)
__global__ void gemm_bf16(const __bf16* __restrict__ A,
                          const __bf16* __restrict__ Wt,
                          const float* __restrict__ bias,
                          void* __restrict__ Cout, int M, int N, int K) {
  constexpr int BM = 128, BN = 128, BK = 32, AP = BK + 8;  // pitch 40 via TDM pad
  __shared__ __bf16 As[2][BM][AP];
  __shared__ __bf16 Ws[2][BN][AP];

  const int tid  = threadIdx.x;
  const int lane = tid & 31, wave = tid >> 5;
  const int wm = wave >> 1, wn = wave & 1;       // 4 x 2 wave grid
  const int m0 = blockIdx.y * BM;
  const int n0 = blockIdx.x * BN;
  const __bf16* Ab = A  + (size_t)m0 * K;
  const __bf16* Wb = Wt + (size_t)n0 * K;

  v8f acc[2][4] = {};

  const int nk = K / BK;
  // Row of 32 bf16 = 16 dwords -> pad_interval code 3; pad 4 dwords -> code 3.
  if (wave == 0) {
    tdm_load_2d_bf16(lds_addr_of(&As[0][0][0]), Ab, BK, BM, K, 3, 3);
    tdm_load_2d_bf16(lds_addr_of(&Ws[0][0][0]), Wb, BK, BN, K, 3, 3);
  }
  for (int kt = 0; kt < nk; ++kt) {
    const int cur = kt & 1;
    if (wave == 0) {
      if (kt + 1 < nk) {
        tdm_load_2d_bf16(lds_addr_of(&As[cur ^ 1][0][0]), Ab + (kt + 1) * BK,
                         BK, BM, K, 3, 3);
        tdm_load_2d_bf16(lds_addr_of(&Ws[cur ^ 1][0][0]), Wb + (kt + 1) * BK,
                         BK, BN, K, 3, 3);
        __builtin_amdgcn_s_wait_tensorcnt(2);   // current buffer complete
      } else {
        __builtin_amdgcn_s_wait_tensorcnt(0);
      }
    }
    __syncthreads();

    const v16bf a0 = lds_frag_a(&As[cur][0][0], AP, wm * 32 + 0,  0);
    const v16bf a1 = lds_frag_a(&As[cur][0][0], AP, wm * 32 + 16, 0);
    v16bf bfr[4];
#pragma unroll
    for (int j = 0; j < 4; ++j)
      bfr[j] = lds_frag_b(&Ws[cur][0][0], AP, wn * 64 + j * 16, 0);
#pragma unroll
    for (int j = 0; j < 4; ++j) {
      acc[0][j] = __builtin_amdgcn_wmma_f32_16x16x32_bf16(false, a0, false,
                              bfr[j], (short)0, acc[0][j], false, false);
      acc[1][j] = __builtin_amdgcn_wmma_f32_16x16x32_bf16(false, a1, false,
                              bfr[j], (short)0, acc[1][j], false, false);
    }
    __syncthreads();
  }

  // Epilogue: C-layout row = r + 8*(lane>=16), col = lane&15.
  const int grp = lane >> 4, col0 = lane & 15;
#pragma unroll
  for (int i = 0; i < 2; ++i)
#pragma unroll
    for (int j = 0; j < 4; ++j) {
      const int col = n0 + wn * 64 + j * 16 + col0;
      const float bv = bias[col];
#pragma unroll
      for (int r = 0; r < 8; ++r) {
        const int row = m0 + wm * 32 + i * 16 + r + 8 * grp;
        float v = acc[i][j][r] + bv;
        if (RELU) v = fmaxf(v, 0.f);
        if (OUT_BF16)
          ((__bf16*)Cout)[(size_t)row * N + col] = (__bf16)v;
        else
          ((float*)Cout)[(size_t)row * N + col] = v;
      }
    }
}

// ---------------------------------------------------------------------------
// Flash attention (bf16 in / bf16 out).  Per (b,h): q/k/v are contiguous
// [S,DH] slabs of the QKV output; out written in (B,H,S,DH) order == flat
// [4096,1024] for the proj GEMM.  128 q rows/block (16 per wave), 32-key
// tiles, online softmax.  Q + K tiles staged by TDM (K double-buffered);
// V is staged manually transposed (TDM cannot transpose).
// ---------------------------------------------------------------------------
__launch_bounds__(256)
__global__ void attention_kernel(const __bf16* __restrict__ qkv,
                                 __bf16* __restrict__ out) {
  constexpr int QT = 128, KT = 32, QP = DH_ + 8, VP = KT + 8;
  __shared__ __bf16 Qs[QT][QP];        // [q][dh]     (pitch 72 via TDM pad)
  __shared__ __bf16 Ks[2][KT][QP];     // [kv][dh]    (B layout [n][k] for QK^T)
  __shared__ __bf16 Vts[DH_][VP];      // [dh][kv]    (B layout [n][k] for P@V)
  __shared__ __bf16 Ps[8][16][VP];     // per-wave P tile [q][kv]

  const int tid = threadIdx.x, lane = tid & 31, wave = tid >> 5;
  const int qt = blockIdx.x, h = blockIdx.y, b = blockIdx.z;

  const __bf16* qp = qkv + (size_t)b * S_ * 3 * E_ + (size_t)h * S_ * DH_;
  const __bf16* kp = qp + (size_t)H_ * S_ * DH_;
  const __bf16* vp = kp + (size_t)H_ * S_ * DH_;
  __bf16* op = out + (size_t)b * H_ * S_ * DH_ + (size_t)h * S_ * DH_;

  const int q0 = qt * QT;
  const int ntiles = (q0 + QT) / KT;   // causal: keys <= last query row

  // Row of 64 bf16 = 32 dwords -> pad_interval code 4; pad 4 dwords -> code 3.
  if (wave == 0) {
    tdm_load_2d_bf16(lds_addr_of(&Qs[0][0]), qp + (size_t)q0 * DH_,
                     DH_, QT, DH_, 4, 3);
    tdm_load_2d_bf16(lds_addr_of(&Ks[0][0][0]), kp, DH_, KT, DH_, 4, 3);
  }

  const int qrow0 = q0 + wave * 16;
  const int grp = lane >> 4, col0 = lane & 15;
  v8f o[4] = {};
  float m_run[8], l_run[8];
#pragma unroll
  for (int r = 0; r < 8; ++r) { m_run[r] = -__builtin_inff(); l_run[r] = 0.f; }

  for (int t = 0; t < ntiles; ++t) {
    const int kv0 = t * KT;
    const int cur = t & 1;
    if (wave == 0) {
      if (t + 1 < ntiles) {
        tdm_load_2d_bf16(lds_addr_of(&Ks[cur ^ 1][0][0]),
                         kp + (size_t)(kv0 + KT) * DH_, DH_, KT, DH_, 4, 3);
        __builtin_amdgcn_s_wait_tensorcnt(1);
      } else {
        __builtin_amdgcn_s_wait_tensorcnt(0);
      }
    }
    // Stage V tile transposed [dh][kv] (manual; overlapped with TDM wait).
    {
      const int r = tid >> 3, c0 = (tid & 7) * 8;     // 32 rows x 8 chunks
      uint4 d = *(const uint4*)(vp + (size_t)(kv0 + r) * DH_ + c0);
      __bf16 tmp[8];
      __builtin_memcpy(tmp, &d, 16);
#pragma unroll
      for (int i = 0; i < 8; ++i) Vts[c0 + i][r] = tmp[i];
    }
    __syncthreads();

    // scores 16x32 = Q[16x64] @ K^T[64x32], two K-steps of 32.
    v8f s[2] = {};
#pragma unroll
    for (int ks = 0; ks < 2; ++ks) {
      const v16bf a = lds_frag_a(&Qs[0][0], QP, wave * 16, ks * 32);
#pragma unroll
      for (int j = 0; j < 2; ++j) {
        const v16bf bb = lds_frag_b(&Ks[cur][0][0], QP, j * 16, ks * 32);
        s[j] = __builtin_amdgcn_wmma_f32_16x16x32_bf16(false, a, false, bb,
                                                (short)0, s[j], false, false);
      }
    }

    // Scale 1/sqrt(E) = 1/32 and causal mask.
#pragma unroll
    for (int j = 0; j < 2; ++j) {
      const int col = kv0 + j * 16 + col0;
#pragma unroll
      for (int r = 0; r < 8; ++r) {
        const int row = qrow0 + r + 8 * grp;
        const float v = s[j][r] * 0.03125f;
        s[j][r] = (col <= row) ? v : -__builtin_inff();
      }
    }

    // Online softmax: 16-lane row reductions (rows live in 16-lane halves).
    float p0[8], p1[8];
#pragma unroll
    for (int r = 0; r < 8; ++r) {
      float mx = fmaxf(s[0][r], s[1][r]);
#pragma unroll
      for (int off = 8; off > 0; off >>= 1) mx = fmaxf(mx, __shfl_xor(mx, off, 32));
      const float m_new = fmaxf(m_run[r], mx);
      const float alpha = __expf(m_run[r] - m_new);
      const float e0 = __expf(s[0][r] - m_new);
      const float e1 = __expf(s[1][r] - m_new);
      float rs = e0 + e1;
#pragma unroll
      for (int off = 8; off > 0; off >>= 1) rs += __shfl_xor(rs, off, 32);
      l_run[r] = l_run[r] * alpha + rs;
      m_run[r] = m_new;
      p0[r] = e0; p1[r] = e1;
#pragma unroll
      for (int j4 = 0; j4 < 4; ++j4) o[j4][r] *= alpha;
    }

    // P (C layout) -> per-wave LDS slab [q][kv] -> reload as A operand.
#pragma unroll
    for (int r = 0; r < 8; ++r) {
      Ps[wave][r + 8 * grp][col0]      = (__bf16)p0[r];
      Ps[wave][r + 8 * grp][16 + col0] = (__bf16)p1[r];
    }
    __syncthreads();

    const v16bf pa = lds_frag_a(&Ps[wave][0][0], VP, 0, 0);
#pragma unroll
    for (int j4 = 0; j4 < 4; ++j4) {
      const v16bf bb = lds_frag_b(&Vts[0][0], VP, j4 * 16, 0);
      o[j4] = __builtin_amdgcn_wmma_f32_16x16x32_bf16(false, pa, false, bb,
                                               (short)0, o[j4], false, false);
    }
    __syncthreads();
  }

  // Normalize and write out[b,h,q,dh] (bf16 for the proj GEMM).
#pragma unroll
  for (int r = 0; r < 8; ++r) {
    const int row = qrow0 + r + 8 * grp;
    const float inv = 1.f / l_run[r];
#pragma unroll
    for (int j4 = 0; j4 < 4; ++j4)
      op[(size_t)row * DH_ + j4 * 16 + col0] = (__bf16)(o[j4][r] * inv);
  }
}

// ---------------------------------------------------------------------------
// Residual add + LayerNorm (fp32 math, bf16 output for the FFN GEMM):
//   out = (hproj + x - mu) * rsqrt(var+eps) * g + b
// ---------------------------------------------------------------------------
__launch_bounds__(256)
__global__ void residual_ln(const float* __restrict__ hp,
                            const float* __restrict__ x,
                            const float* __restrict__ g,
                            const float* __restrict__ beta,
                            __bf16* __restrict__ out) {
  __shared__ float red[256];
  const int row = blockIdx.x;
  const int tid = threadIdx.x;
  const float* hr = hp + (size_t)row * E_;
  const float* xr = x  + (size_t)row * E_;

  float v[4];
  float s = 0.f;
#pragma unroll
  for (int i = 0; i < 4; ++i) {
    v[i] = hr[tid + i * 256] + xr[tid + i * 256];
    s += v[i];
  }
  red[tid] = s;
  __syncthreads();
  for (int w = 128; w > 0; w >>= 1) {
    if (tid < w) red[tid] += red[tid + w];
    __syncthreads();
  }
  const float mu = red[0] * (1.f / E_);
  __syncthreads();

  float sq = 0.f;
#pragma unroll
  for (int i = 0; i < 4; ++i) { const float d = v[i] - mu; sq += d * d; }
  red[tid] = sq;
  __syncthreads();
  for (int w = 128; w > 0; w >>= 1) {
    if (tid < w) red[tid] += red[tid + w];
    __syncthreads();
  }
  const float rinv = rsqrtf(red[0] * (1.f / E_) + LN_EPS);

#pragma unroll
  for (int i = 0; i < 4; ++i) {
    const int c = tid + i * 256;
    out[(size_t)row * E_ + c] = (__bf16)((v[i] - mu) * rinv * g[c] + beta[c]);
  }
}

// ---------------------------------------------------------------------------
extern "C" void kernel_launch(void* const* d_in, const int* in_sizes, int n_in,
                              void* d_out, int out_size, void* d_ws,
                              size_t ws_size, hipStream_t stream) {
  const float* x      = (const float*)d_in[0];
  const float* w_qkv  = (const float*)d_in[1];
  const float* b_qkv  = (const float*)d_in[2];
  const float* w_proj = (const float*)d_in[3];
  const float* b_proj = (const float*)d_in[4];
  const float* ln_g   = (const float*)d_in[5];
  const float* ln_b   = (const float*)d_in[6];
  const float* w1     = (const float*)d_in[7];
  const float* b1     = (const float*)d_in[8];
  const float* w2     = (const float*)d_in[9];
  const float* b2     = (const float*)d_in[10];
  float* out = (float*)d_out;

  // Workspace layout (bytes); ff1 reuses the dead qkv+attn region. Peak ~88MB.
  char* p = (char*)d_ws;
  __bf16* xb     = (__bf16*)p;  p += (size_t)M_TOT * E_ * 2;        // 8 MB
  __bf16* wqkvT  = (__bf16*)p;  p += (size_t)3 * E_ * E_ * 2;       // 6 MB
  __bf16* wprojT = (__bf16*)p;  p += (size_t)E_ * E_ * 2;           // 2 MB
  __bf16* w1T    = (__bf16*)p;  p += (size_t)FF_ * E_ * 2;          // 8 MB
  __bf16* w2T    = (__bf16*)p;  p += (size_t)E_ * FF_ * 2;          // 8 MB
  __bf16* qkvb   = (__bf16*)p;  p += (size_t)M_TOT * 3 * E_ * 2;    // 24 MB
  __bf16* attnb  = (__bf16*)p;  p += (size_t)M_TOT * E_ * 2;        // 8 MB
  float*  projf  = (float*)p;   p += (size_t)M_TOT * E_ * 4;        // 16 MB
  __bf16* lnb    = (__bf16*)p;  p += (size_t)M_TOT * E_ * 2;        // 8 MB
  __bf16* ff1b   = qkvb;        // 32 MB alias over qkv+attn (dead by then)

  const dim3 blk(256);
  // One-time conversions / weight transposes.
  convert_bf16<<<dim3(M_TOT * E_ / 1024), blk, 0, stream>>>(x, xb, M_TOT * E_);
  transpose_to_bf16<<<dim3(3 * E_ / 32, E_ / 32), blk, 0, stream>>>(w_qkv, wqkvT, E_, 3 * E_);
  transpose_to_bf16<<<dim3(E_ / 32, E_ / 32), blk, 0, stream>>>(w_proj, wprojT, E_, E_);
  transpose_to_bf16<<<dim3(FF_ / 32, E_ / 32), blk, 0, stream>>>(w1, w1T, E_, FF_);
  transpose_to_bf16<<<dim3(E_ / 32, FF_ / 32), blk, 0, stream>>>(w2, w2T, FF_, E_);

  gemm_bf16<0, 1><<<dim3(3 * E_ / 128, M_TOT / 128), blk, 0, stream>>>(
      xb, wqkvT, b_qkv, qkvb, M_TOT, 3 * E_, E_);
  attention_kernel<<<dim3(S_ / 128, H_, B_), blk, 0, stream>>>(qkvb, attnb);
  gemm_bf16<0, 0><<<dim3(E_ / 128, M_TOT / 128), blk, 0, stream>>>(
      attnb, wprojT, b_proj, projf, M_TOT, E_, E_);
  residual_ln<<<dim3(M_TOT), blk, 0, stream>>>(projf, x, ln_g, ln_b, lnb);
  gemm_bf16<1, 1><<<dim3(FF_ / 128, M_TOT / 128), blk, 0, stream>>>(
      lnb, w1T, b1, ff1b, M_TOT, FF_, E_);
  gemm_bf16<0, 0><<<dim3(E_ / 128, M_TOT / 128), blk, 0, stream>>>(
      ff1b, w2T, b2, out, M_TOT, E_, FF_);
}
